// SEGNNModel_31825707663896
// MI455X (gfx1250) — matrix operand order repaired and checked by reading, not compiled, not run on hardware
//
#include <hip/hip_runtime.h>
#include <cmath>

// ============================================================================
// SEGNN forward for MI455X (gfx1250, wave32, WMMA).
//
// All attr scalars in the reference are 1.0, so each O(3) tensor product is 5
// dense matmuls over shared activation rows. We tile 16 edges/nodes per wave,
// stage activations in LDS as f16 *already in WMMA A-fragment order*, and
// pre-swizzle the f16 weights into B-fragment order at conversion time, so the
// inner loop is: 1 v16h LDS load (A) + 1 v16h coalesced global load (B) +
// v_wmma_f32_16x16x32_f16. ~35 GFLOP vs ~350MB traffic at 23.3 TB/s: WMMA-f16
// balances compute with HBM; scalar f32 VALU would be >10x compute bound.
//
// Input flattening assumption (setup_inputs() dict insertion order, recursive):
//  0:x 1:pos 2:edge_index 3:batch
//  emb1: 4 w_ss(10x96) 5 b(96) 6 w_sv(10x32)
//  emb2: 7 w_ss 8 b 9 w_vv2s 10 w_sv 11 w_vs2v 12 w_vxv
//  layer0: m1 13..18  m2 19..24  u1 25..30  u2 31..36
//  layer1: m1 37..42  m2 43..48  u1 49..54  u2 55..60
//  pre1: 61 w_ss(64x128) 62 b(128) 63 w_vv2s(32x128)
//  pre2: 64 w(128x128) 65 b(128)
//  post1: 66 w(128x128) 67 b(128)
//  post2: 68 w(128x1) 69 b(1)
// ============================================================================

typedef __attribute__((ext_vector_type(16))) _Float16 v16h;
typedef __attribute__((ext_vector_type(8)))  float    v8f;

__device__ __forceinline__ float sigm_(float x) { return 1.0f / (1.0f + __expf(-x)); }
__device__ __forceinline__ float silu_(float x) { return x * sigm_(x); }

// ---------------------------------------------------------------------------
// Fragment packing (CDNA5 ISA §7.12.2, 16-bit A 16x32 / B 32x16).
// A-fragment storage: [ktile][lane(32)][16 halves]; lane = (m&15) + 16*h.
// Inverse of: K = (j<4 ? h*8+2j : 16+h*8+2(j-4)) + (e&1), j = e>>1.
__device__ __forceinline__ int a_pack_idx(int m, int k) {
  const int kt = k >> 5;
  const int kk = k & 31;
  const int g  = kk >> 4;         // 0: K%32 in [0,16) -> VGPR 0..3 ; 1: VGPR 4..7
  const int h  = (kk >> 3) & 1;   // lane half
  const int rr = kk & 7;
  const int e  = ((g << 2) + (rr >> 1)) * 2 + (rr & 1);
  return ((kt << 5) + (m & 15) + (h << 4)) * 16 + e;
}
// B-fragment storage: [ntile][ktile][lane(32)][16 halves]; lane=(n&15)+16*h,
// element e holds K = kt*32 + (e<8 ? h*8+e : 16+h*8+(e-8)), N = nt*16 + (n&15).
__device__ __forceinline__ int b_pack_idx(int k, int n, int KT) {
  return ((((n >> 4) * KT + (k >> 5)) << 5) + (n & 15) + (((k >> 3) & 1) << 4)) * 16 + (((k >> 4) & 1) << 3) + (k & 7);
}

__device__ __forceinline__ v16h load_frag(const _Float16* p, int fragIdx, int lane) {
  return *(const v16h*)(p + (size_t)((fragIdx << 5) + lane) * 16);
}
__device__ __forceinline__ v8f wmma32(v16h a, v16h b, v8f c) {
  return __builtin_amdgcn_wmma_f32_16x16x32_f16(false, a, false, b, (short)0, c, false, false);
}

// ---------- device/host shared TP descriptor ---------------------------------
struct TPDev {
  const _Float16* wcat;   // packed B frags: [w_ss | w_sv], K=A0PAD, N=B0+B1
  const _Float16* wv2s;   // packed B frags: w_vv2s, K=A1PAD, N=B0
  const _Float16* wvcat;  // packed B frags: [w_vs2v ; w_vxv], K=2*A1PAD, N=B1
  const float*    bias;   // [B0] (f32)
  float rs, rv;           // 1/sqrt(fan_s), 1/sqrt(fan_v)
};

// ---------- wave-level 16-row tensor product ---------------------------------
// SinF: LDS f16, A-fragment-packed, 16*A0PAD halves (zero padded past real a0)
// XV  : LDS f32 [16][A1][3], AV : LDS f32 [16][3]
// DTF : LDS f16 scratch (A-fragment-packed), >= 16*2*A1PAD halves
// Sout: LDS f32 [16][B0], Vout: LDS f32 [16][B1][3]
template <int A0PAD, int A1, int A1PAD, int B0, int B1>
__device__ void tile_tp(TPDev d, int lane,
                        const _Float16* SinF, const float* XV, const float* AV,
                        _Float16* DTF, float* Sout, float* Vout) {
  constexpr int BN  = B0 + B1;
  constexpr int NT  = BN / 16;
  constexpr int KT0 = A0PAD / 32;
  const v8f zero8 = {0.f, 0.f, 0.f, 0.f, 0.f, 0.f, 0.f, 0.f};
  v8f acc[NT];
#pragma unroll
  for (int t = 0; t < NT; ++t) acc[t] = zero8;

  // scalar path: [16,A0PAD] x [A0PAD,BN]
#pragma unroll 1
  for (int kt = 0; kt < KT0; ++kt) {
    v16h a = load_frag(SinF, kt, lane);
#pragma unroll
    for (int t = 0; t < NT; ++t) {
      v16h b = load_frag(d.wcat, t * KT0 + kt, lane);
      acc[t] = wmma32(a, b, acc[t]);
    }
  }

  if constexpr (A1 > 0) {
    constexpr int KT1 = A1PAD / 32;
    // dot(xv, av) tile -> DTF (packed), then @ w_vv2s into the B0 columns
    for (int idx = lane; idx < 16 * A1PAD; idx += 32) {
      int m = idx / A1PAD, a = idx % A1PAD;
      float val = 0.f;
      if (a < A1) {
        const float* xv = XV + (m * A1 + a) * 3;
        const float* av = AV + m * 3;
        val = xv[0] * av[0] + xv[1] * av[1] + xv[2] * av[2];
      }
      DTF[a_pack_idx(m, a)] = (_Float16)val;
    }
    __syncthreads();
#pragma unroll 1
    for (int kt = 0; kt < KT1; ++kt) {
      v16h a = load_frag(DTF, kt, lane);
#pragma unroll
      for (int t = 0; t < B0 / 16; ++t) {
        v16h b = load_frag(d.wv2s, t * KT1 + kt, lane);
        acc[t] = wmma32(a, b, acc[t]);
      }
    }
    __syncthreads();
  }

  // write scalar output: C layout lane->(M=r+8h, N=lane&15)
  const int n = lane & 15, hh = (lane >> 4) & 1;
#pragma unroll
  for (int t = 0; t < B0 / 16; ++t) {
#pragma unroll
    for (int r = 0; r < 8; ++r) {
      int m = r + 8 * hh;
      Sout[m * B0 + t * 16 + n] = acc[t][r] * d.rs + d.bias[t * 16 + n];
    }
  }

  if constexpr (B1 > 0) {
    constexpr int VT = B1 / 16;
#pragma unroll 1
    for (int dd = 0; dd < 3; ++dd) {
      v8f vacc[VT];
#pragma unroll
      for (int t = 0; t < VT; ++t) {
#pragma unroll
        for (int r = 0; r < 8; ++r) {
          int m = r + 8 * hh;
          vacc[t][r] = acc[B0 / 16 + t][r] * AV[m * 3 + dd];  // (xs@w_sv)*av_d
        }
      }
      if constexpr (A1 > 0) {
        constexpr int K2  = 2 * A1PAD;
        constexpr int KT2 = K2 / 32;
        __syncthreads();
        for (int idx = lane; idx < 16 * K2; idx += 32) {
          int m = idx / K2, a = idx % K2;
          float val = 0.f;
          if (a < A1) {
            val = XV[(m * A1 + a) * 3 + dd];
          } else if (a >= A1PAD && a < A1PAD + A1) {
            const float* xv = XV + (m * A1 + (a - A1PAD)) * 3;
            const float* av = AV + m * 3;
            const int i1 = (dd + 1) % 3, i2 = (dd + 2) % 3;
            val = xv[i1] * av[i2] - xv[i2] * av[i1];  // (xv x av)_dd
          }
          DTF[a_pack_idx(m, a)] = (_Float16)val;
        }
        __syncthreads();
#pragma unroll 1
        for (int kt = 0; kt < KT2; ++kt) {
          v16h a = load_frag(DTF, kt, lane);
#pragma unroll
          for (int t = 0; t < VT; ++t) {
            v16h b = load_frag(d.wvcat, t * KT2 + kt, lane);
            vacc[t] = wmma32(a, b, vacc[t]);
          }
        }
      }
#pragma unroll
      for (int t = 0; t < VT; ++t) {
#pragma unroll
        for (int r = 0; r < 8; ++r) {
          int m = r + 8 * hh;
          Vout[(m * B1 + t * 16 + n) * 3 + dd] = vacc[t][r] * d.rv;
        }
      }
    }
  }
}

// ============================ utility kernels ================================
__global__ void zero_f32(float* p, int n) {
  for (int i = blockIdx.x * blockDim.x + threadIdx.x; i < n; i += gridDim.x * blockDim.x) p[i] = 0.f;
}
// Packed-B conversion: dst holds [A | B] (rows<K, zero-pad to Kpad) in
// B-fragment order for KT = Kpad/32.
__global__ void cvt_cat(const float* __restrict__ A, int NA, const float* __restrict__ B, int NB,
                        _Float16* __restrict__ dst, int K, int Kpad) {
  const int N = NA + NB, total = Kpad * N, KT = Kpad >> 5;
  for (int i = blockIdx.x * blockDim.x + threadIdx.x; i < total; i += gridDim.x * blockDim.x) {
    int k = i / N, n = i % N;
    float v = 0.f;
    if (k < K) v = (n < NA) ? A[k * NA + n] : B[k * NB + (n - NA)];
    dst[b_pack_idx(k, n, KT)] = (_Float16)v;
  }
}
// Packed-B conversion: logical rows [0,K)=A, [Kpad,Kpad+K)=B, zeros elsewhere,
// total logical K = 2*Kpad, fragment order with KT = 2*Kpad/32.
__global__ void cvt_stack(const float* __restrict__ A, const float* __restrict__ B,
                          _Float16* __restrict__ dst, int K, int N, int Kpad) {
  const int total = 2 * Kpad * N, KT = (2 * Kpad) >> 5;
  for (int i = blockIdx.x * blockDim.x + threadIdx.x; i < total; i += gridDim.x * blockDim.x) {
    int k = i / N, n = i % N;
    float v = 0.f;
    if (k < K) v = A[k * N + n];
    else if (k >= Kpad && k < Kpad + K) v = B[(k - Kpad) * N + n];
    dst[b_pack_idx(k, n, KT)] = (_Float16)v;
  }
}

// edge geometry + node-attr accumulation
__global__ void edge_geom(const float* __restrict__ pos, const int* __restrict__ ei, int E,
                          float* __restrict__ ea, float* __restrict__ edist,
                          float* __restrict__ nav, float* __restrict__ deg) {
  int e = blockIdx.x * blockDim.x + threadIdx.x;
  if (e >= E) return;
  int sI = ei[e], rI = ei[E + e];
  float rx = pos[sI * 3 + 0] - pos[rI * 3 + 0];
  float ry = pos[sI * 3 + 1] - pos[rI * 3 + 1];
  float rz = pos[sI * 3 + 2] - pos[rI * 3 + 2];
  float dd = rx * rx + ry * ry + rz * rz;
  float rn = sqrtf(fmaxf(dd, 1e-12f));
  const float s3 = 1.7320508075688772f;
  float ex = s3 * rx / rn, ey = s3 * ry / rn, ez = s3 * rz / rn;
  ea[e * 3 + 0] = ex; ea[e * 3 + 1] = ey; ea[e * 3 + 2] = ez;
  edist[e] = dd;
  atomicAdd(&nav[rI * 3 + 0], ex);
  atomicAdd(&nav[rI * 3 + 1], ey);
  atomicAdd(&nav[rI * 3 + 2], ez);
  atomicAdd(&deg[rI], 1.f);
}
__global__ void na_fin(float* __restrict__ nav, const float* __restrict__ deg, int N) {
  int i = blockIdx.x * blockDim.x + threadIdx.x;
  if (i >= N) return;
  float d = fmaxf(deg[i], 1.f);
  nav[i * 3 + 0] /= d; nav[i * 3 + 1] /= d; nav[i * 3 + 2] /= d;
}

// ============================ embedding ======================================
__global__ void __launch_bounds__(32)
embed_kernel(const int* __restrict__ x, const float* __restrict__ nav, int N,
             const float* __restrict__ e1_wss, const float* __restrict__ e1_b,
             const float* __restrict__ e1_wsv, TPDev e2,
             float* __restrict__ sN, float* __restrict__ vN) {
  __shared__ float    S1[16 * 96];
  __shared__ _Float16 SinF[16 * 64];
  __shared__ float    XV[16 * 32 * 3];
  __shared__ float    AV[16 * 3];
  __shared__ _Float16 DTF[16 * 64];
  __shared__ float    Sout[16 * 64];
  __shared__ float    Vout[16 * 32 * 3];
  __shared__ int      XI[16];
  const int lane = threadIdx.x;
  const int n0 = blockIdx.x * 16;
  if (lane < 16) {
    int n = n0 + lane; if (n >= N) n = N - 1;
    XI[lane] = x[n];
    AV[lane * 3 + 0] = nav[n * 3 + 0];
    AV[lane * 3 + 1] = nav[n * 3 + 1];
    AV[lane * 3 + 2] = nav[n * 3 + 2];
  }
  __syncthreads();
  const float r10 = 0.31622776601683794f;  // 1/sqrt(10)
  for (int idx = lane; idx < 16 * 96; idx += 32) {
    int m = idx / 96, c = idx % 96;
    S1[idx] = e1_wss[XI[m] * 96 + c] * r10 + e1_b[c];
  }
  __syncthreads();
  for (int idx = lane; idx < 16 * 64; idx += 32) {
    int m = idx >> 6, c = idx & 63;
    SinF[a_pack_idx(m, c)] = (_Float16)silu_(S1[m * 96 + c]);
  }
  for (int idx = lane; idx < 16 * 32 * 3; idx += 32) {
    int m = idx / 96, r = idx % 96, j = r / 3, dd = r % 3;
    XV[idx] = (e1_wsv[XI[m] * 32 + j] * AV[m * 3 + dd] * r10) * sigm_(S1[m * 96 + 64 + j]);
  }
  __syncthreads();
  tile_tp<64, 32, 32, 64, 32>(e2, lane, SinF, XV, AV, DTF, Sout, Vout);
  __syncthreads();
  for (int idx = lane; idx < 16 * 64; idx += 32) {
    int m = idx >> 6, c = idx & 63, n = n0 + m;
    if (n < N) sN[n * 64 + c] = Sout[idx];
  }
  for (int idx = lane; idx < 16 * 96; idx += 32) {
    int m = idx / 96, r = idx % 96, n = n0 + m;
    if (n < N) vN[n * 96 + r] = Vout[idx];
  }
}

// ============================ per-edge messages ==============================
__global__ void __launch_bounds__(32)
msg_kernel(const float* __restrict__ sN, const float* __restrict__ vN,
           const int* __restrict__ ei, int E,
           const float* __restrict__ ea, const float* __restrict__ edist,
           TPDev m1, TPDev m2,
           float* __restrict__ ags, float* __restrict__ agv) {
  __shared__ _Float16 Sin1F[16 * 160];
  __shared__ float    XV1[16 * 64 * 3];
  __shared__ float    AV[16 * 3];
  __shared__ _Float16 DTF[16 * 128];
  __shared__ float    Sout[16 * 96];
  __shared__ float    Vout[16 * 32 * 3];
  __shared__ _Float16 Sin2F[16 * 64];
  __shared__ float    XV2[16 * 32 * 3];
  __shared__ int      RECV[16], SEND[16];
  const int lane = threadIdx.x;
  const int e0 = blockIdx.x * 16;
  if (lane < 16) {
    int e = e0 + lane; if (e >= E) e = E - 1;
    SEND[lane] = ei[e];
    RECV[lane] = ei[E + e];
    AV[lane * 3 + 0] = ea[e * 3 + 0];
    AV[lane * 3 + 1] = ea[e * 3 + 1];
    AV[lane * 3 + 2] = ea[e * 3 + 2];
  }
  __syncthreads();
  // Sin1: [s_recv(64) | s_send(64) | d2(1) | pad(31)]
  for (int idx = lane; idx < 16 * 160; idx += 32) {
    int m = idx / 160, c = idx % 160;
    float val = 0.f;
    if (c < 64)        val = sN[RECV[m] * 64 + c];
    else if (c < 128)  val = sN[SEND[m] * 64 + (c - 64)];
    else if (c == 128) { int e = e0 + m; if (e >= E) e = E - 1; val = edist[e]; }
    Sin1F[a_pack_idx(m, c)] = (_Float16)val;
  }
  // XV1: [v_recv(32) | v_send(32)] x 3
  for (int idx = lane; idx < 16 * 64 * 3; idx += 32) {
    int m = idx / 192, r = idx % 192, a = r / 3, dd = r % 3;
    XV1[idx] = (a < 32) ? vN[RECV[m] * 96 + a * 3 + dd]
                        : vN[SEND[m] * 96 + (a - 32) * 3 + dd];
  }
  __syncthreads();
  tile_tp<160, 64, 64, 96, 32>(m1, lane, Sin1F, XV1, AV, DTF, Sout, Vout);
  __syncthreads();
  // gate -> inputs of m2
  for (int idx = lane; idx < 16 * 64; idx += 32) {
    int m = idx >> 6, c = idx & 63;
    Sin2F[a_pack_idx(m, c)] = (_Float16)silu_(Sout[m * 96 + c]);
  }
  for (int idx = lane; idx < 16 * 32 * 3; idx += 32) {
    int m = idx / 96, j = (idx % 96) / 3;
    XV2[idx] = Vout[idx] * sigm_(Sout[m * 96 + 64 + j]);
  }
  __syncthreads();
  tile_tp<64, 32, 32, 96, 32>(m2, lane, Sin2F, XV2, AV, DTF, Sout, Vout);
  __syncthreads();
  // gate + scatter-add to targets
  for (int idx = lane; idx < 16 * 64; idx += 32) {
    int m = idx >> 6, c = idx & 63;
    if (e0 + m < E) atomicAdd(&ags[RECV[m] * 64 + c], silu_(Sout[m * 96 + c]));
  }
  for (int idx = lane; idx < 16 * 32 * 3; idx += 32) {
    int m = idx / 96, r = idx % 96, j = r / 3;
    if (e0 + m < E) atomicAdd(&agv[RECV[m] * 96 + r], Vout[idx] * sigm_(Sout[m * 96 + 64 + j]));
  }
}

// ============================ per-node update ================================
__global__ void __launch_bounds__(32)
upd_kernel(float* __restrict__ sN, float* __restrict__ vN,
           const float* __restrict__ ags, const float* __restrict__ agv,
           const float* __restrict__ nav, int N, TPDev u1, TPDev u2) {
  __shared__ _Float16 Sin1F[16 * 128];
  __shared__ float    XV1[16 * 64 * 3];
  __shared__ float    AV[16 * 3];
  __shared__ _Float16 DTF[16 * 128];
  __shared__ float    Sout[16 * 96];
  __shared__ float    Vout[16 * 32 * 3];
  __shared__ _Float16 Sin2F[16 * 64];
  __shared__ float    XV2[16 * 32 * 3];
  const int lane = threadIdx.x;
  const int n0 = blockIdx.x * 16;
  if (lane < 16) {
    int n = n0 + lane; if (n >= N) n = N - 1;
    AV[lane * 3 + 0] = nav[n * 3 + 0];
    AV[lane * 3 + 1] = nav[n * 3 + 1];
    AV[lane * 3 + 2] = nav[n * 3 + 2];
  }
  __syncthreads();
  for (int idx = lane; idx < 16 * 128; idx += 32) {
    int m = idx >> 7, c = idx & 127;
    int n = n0 + m; if (n >= N) n = N - 1;
    float val = (c < 64) ? sN[n * 64 + c] : ags[n * 64 + (c - 64)];
    Sin1F[a_pack_idx(m, c)] = (_Float16)val;
  }
  for (int idx = lane; idx < 16 * 64 * 3; idx += 32) {
    int m = idx / 192, r = idx % 192, a = r / 3, dd = r % 3;
    int n = n0 + m; if (n >= N) n = N - 1;
    XV1[idx] = (a < 32) ? vN[n * 96 + a * 3 + dd] : agv[n * 96 + (a - 32) * 3 + dd];
  }
  __syncthreads();
  tile_tp<128, 64, 64, 96, 32>(u1, lane, Sin1F, XV1, AV, DTF, Sout, Vout);
  __syncthreads();
  for (int idx = lane; idx < 16 * 64; idx += 32) {
    int m = idx >> 6, c = idx & 63;
    Sin2F[a_pack_idx(m, c)] = (_Float16)silu_(Sout[m * 96 + c]);
  }
  for (int idx = lane; idx < 16 * 32 * 3; idx += 32) {
    int m = idx / 96, j = (idx % 96) / 3;
    XV2[idx] = Vout[idx] * sigm_(Sout[m * 96 + 64 + j]);
  }
  __syncthreads();
  tile_tp<64, 32, 32, 64, 32>(u2, lane, Sin2F, XV2, AV, DTF, Sout, Vout);
  __syncthreads();
  // residual
  for (int idx = lane; idx < 16 * 64; idx += 32) {
    int m = idx >> 6, c = idx & 63, n = n0 + m;
    if (n < N) sN[n * 64 + c] += Sout[m * 64 + c];
  }
  for (int idx = lane; idx < 16 * 96; idx += 32) {
    int m = idx / 96, r = idx % 96, n = n0 + m;
    if (n < N) vN[n * 96 + r] += Vout[idx];
  }
}

// ============================ head ==========================================
__global__ void __launch_bounds__(32)
head_kernel(const float* __restrict__ sN, const float* __restrict__ vN,
            const float* __restrict__ nav, const int* __restrict__ batch, int N,
            TPDev pre1, const _Float16* __restrict__ pre2w, const float* __restrict__ pre2b,
            float* __restrict__ gacc, float* __restrict__ gcnt) {
  __shared__ _Float16 SinF[16 * 64];
  __shared__ float    XV[16 * 32 * 3];
  __shared__ float    AV[16 * 3];
  __shared__ _Float16 DTF[16 * 64];
  __shared__ float    Sout[16 * 128];
  __shared__ _Float16 H2F[16 * 128];
  __shared__ int      BI[16];
  const int lane = threadIdx.x;
  const int n0 = blockIdx.x * 16;
  if (lane < 16) {
    int n = n0 + lane; if (n >= N) n = N - 1;
    BI[lane] = batch[n];
    AV[lane * 3 + 0] = nav[n * 3 + 0];
    AV[lane * 3 + 1] = nav[n * 3 + 1];
    AV[lane * 3 + 2] = nav[n * 3 + 2];
  }
  __syncthreads();
  for (int idx = lane; idx < 16 * 64; idx += 32) {
    int m = idx >> 6, c = idx & 63;
    int n = n0 + m; if (n >= N) n = N - 1;
    SinF[a_pack_idx(m, c)] = (_Float16)sN[n * 64 + c];
  }
  for (int idx = lane; idx < 16 * 32 * 3; idx += 32) {
    int m = idx / 96, r = idx % 96;
    int n = n0 + m; if (n >= N) n = N - 1;
    XV[idx] = vN[n * 96 + r];
  }
  __syncthreads();
  tile_tp<64, 32, 32, 128, 0>(pre1, lane, SinF, XV, AV, DTF, Sout, nullptr);
  __syncthreads();
  for (int idx = lane; idx < 16 * 128; idx += 32) {
    int m = idx >> 7, c = idx & 127;
    H2F[a_pack_idx(m, c)] = (_Float16)silu_(Sout[idx]);
  }
  __syncthreads();
  // hs @ pre2.w (+ bias) with WMMA (KT=4), then pooled atomically per graph
  const v8f zero8 = {0.f, 0.f, 0.f, 0.f, 0.f, 0.f, 0.f, 0.f};
  v8f acc[8];
#pragma unroll
  for (int t = 0; t < 8; ++t) acc[t] = zero8;
#pragma unroll 1
  for (int kt = 0; kt < 4; ++kt) {
    v16h a = load_frag(H2F, kt, lane);
#pragma unroll
    for (int t = 0; t < 8; ++t) {
      v16h b = load_frag(pre2w, t * 4 + kt, lane);
      acc[t] = wmma32(a, b, acc[t]);
    }
  }
  const int n = lane & 15, hh = (lane >> 4) & 1;
#pragma unroll
  for (int t = 0; t < 8; ++t) {
#pragma unroll
    for (int r = 0; r < 8; ++r) {
      int m = r + 8 * hh;
      if (n0 + m < N) {
        int col = t * 16 + n;
        atomicAdd(&gacc[BI[m] * 128 + col], acc[t][r] + pre2b[col]);
      }
    }
  }
  if (lane < 16 && (n0 + lane) < N) atomicAdd(&gcnt[BI[lane]], 1.f);
}

__global__ void head_final(const float* __restrict__ gacc, const float* __restrict__ gcnt,
                           const float* __restrict__ p1w, const float* __restrict__ p1b,
                           const float* __restrict__ p2w, const float* __restrict__ p2b,
                           float* __restrict__ out, int G) {
  __shared__ float GM[16 * 128];
  __shared__ float T[16 * 128];
  const int t = threadIdx.x;
  for (int idx = t; idx < G * 128; idx += blockDim.x) {
    int g = idx >> 7;
    GM[idx] = gacc[idx] / fmaxf(gcnt[g], 1.f);
  }
  __syncthreads();
  for (int idx = t; idx < G * 128; idx += blockDim.x) {
    int g = idx >> 7, j = idx & 127;
    float a = p1b[j];
    for (int k = 0; k < 128; ++k) a += GM[g * 128 + k] * p1w[k * 128 + j];
    T[idx] = silu_(a);
  }
  __syncthreads();
  if (t < G) {
    float a = p2b[0];
    for (int j = 0; j < 128; ++j) a += T[t * 128 + j] * p2w[j];
    out[t] = a;
  }
}

// ============================ host launch ====================================
struct TPf32 { const float *wss, *b, *wvv2s, *wsv, *wvs2v, *wvxv; };

extern "C" void kernel_launch(void* const* d_in, const int* in_sizes, int n_in,
                              void* d_out, int out_size, void* d_ws, size_t ws_size,
                              hipStream_t stream) {
  (void)n_in; (void)ws_size;
  const int N = in_sizes[0];
  const int E = in_sizes[2] / 2;
  const int G = out_size;  // 16 graphs

  const int*   x     = (const int*)d_in[0];
  const float* pos   = (const float*)d_in[1];
  const int*   ei    = (const int*)d_in[2];
  const int*   batch = (const int*)d_in[3];
  auto Fp = [&](int i) { return (const float*)d_in[i]; };

  TPf32 emb1{Fp(4), Fp(5), nullptr, Fp(6), nullptr, nullptr};
  auto rd = [&](int b) { return TPf32{Fp(b), Fp(b + 1), Fp(b + 2), Fp(b + 3), Fp(b + 4), Fp(b + 5)}; };
  TPf32 emb2 = rd(7);
  TPf32 Lm1[2] = {rd(13), rd(37)};
  TPf32 Lm2[2] = {rd(19), rd(43)};
  TPf32 Lu1[2] = {rd(25), rd(49)};
  TPf32 Lu2[2] = {rd(31), rd(55)};
  TPf32 pre1{Fp(61), Fp(62), Fp(63), nullptr, nullptr, nullptr};
  const float* pre2_w  = Fp(64); const float* pre2_b  = Fp(65);
  const float* post1_w = Fp(66); const float* post1_b = Fp(67);
  const float* post2_w = Fp(68); const float* post2_b = Fp(69);

  // ---- workspace layout ----
  char* wsb = (char*)d_ws;
  size_t off = 0;
  auto allocH = [&](size_t bytes) {
    void* p = wsb + off;
    off = (off + bytes + 255) & ~(size_t)255;
    return p;
  };
  auto grid = [](int n) { return (n + 255) / 256; };

  auto buildTP = [&](const TPf32& p, int a0, int a0pad, int a1, int a1pad, int b0, int b1) {
    TPDev d;
    _Float16* wcat = (_Float16*)allocH(sizeof(_Float16) * (size_t)a0pad * (b0 + b1));
    cvt_cat<<<grid(a0pad * (b0 + b1)), 256, 0, stream>>>(p.wss, b0, b1 ? p.wsv : p.wss,
                                                         b1 ? b1 : 0, wcat, a0, a0pad);
    d.wcat = wcat;
    d.wv2s = nullptr; d.wvcat = nullptr;
    if (a1 > 0) {
      _Float16* wv2s = (_Float16*)allocH(sizeof(_Float16) * (size_t)a1pad * b0);
      cvt_cat<<<grid(a1pad * b0), 256, 0, stream>>>(p.wvv2s, b0, p.wvv2s, 0, wv2s, a1, a1pad);
      d.wv2s = wv2s;
      if (b1 > 0) {
        _Float16* wvcat = (_Float16*)allocH(sizeof(_Float16) * (size_t)2 * a1pad * b1);
        cvt_stack<<<grid(2 * a1pad * b1), 256, 0, stream>>>(p.wvs2v, p.wvxv, wvcat, a1, b1, a1pad);
        d.wvcat = wvcat;
      }
    }
    d.bias = p.b;
    float fs = (float)a0 + (a1 ? (float)a1 : 0.f);
    float fv = (float)a0 + (a1 ? 2.f * (float)a1 : 0.f);
    d.rs = 1.f / sqrtf(fs);
    d.rv = 1.f / sqrtf(fv);
    return d;
  };

  TPDev dEmb2 = buildTP(emb2, 64, 64, 32, 32, 64, 32);
  TPDev dM1[2], dM2[2], dU1[2], dU2[2];
  for (int L = 0; L < 2; ++L) {
    dM1[L] = buildTP(Lm1[L], 129, 160, 64, 64, 96, 32);
    dM2[L] = buildTP(Lm2[L], 64, 64, 32, 32, 96, 32);
    dU1[L] = buildTP(Lu1[L], 128, 128, 64, 64, 96, 32);
    dU2[L] = buildTP(Lu2[L], 64, 64, 32, 32, 64, 32);
  }
  TPDev dPre1 = buildTP(pre1, 64, 64, 32, 32, 128, 0);
  _Float16* pre2w16 = (_Float16*)allocH(sizeof(_Float16) * 128 * 128);
  cvt_cat<<<grid(128 * 128), 256, 0, stream>>>(pre2_w, 128, pre2_w, 0, pre2w16, 128, 128);

  float* ea   = (float*)allocH(sizeof(float) * (size_t)E * 3);
  float* dis  = (float*)allocH(sizeof(float) * (size_t)E);
  float* deg  = (float*)allocH(sizeof(float) * (size_t)N);
  float* nav  = (float*)allocH(sizeof(float) * (size_t)N * 3);
  float* sN   = (float*)allocH(sizeof(float) * (size_t)N * 64);
  float* vN   = (float*)allocH(sizeof(float) * (size_t)N * 96);
  float* ags  = (float*)allocH(sizeof(float) * (size_t)N * 64);
  float* agv  = (float*)allocH(sizeof(float) * (size_t)N * 96);
  float* gacc = (float*)allocH(sizeof(float) * 16 * 128);
  float* gcnt = (float*)allocH(sizeof(float) * 16);

  const int etiles = (E + 15) / 16;
  const int ntiles = (N + 15) / 16;

  // ---- pipeline ----
  zero_f32<<<grid(N), 256, 0, stream>>>(deg, N);
  zero_f32<<<grid(3 * N), 256, 0, stream>>>(nav, 3 * N);
  edge_geom<<<grid(E), 256, 0, stream>>>(pos, ei, E, ea, dis, nav, deg);
  na_fin<<<grid(N), 256, 0, stream>>>(nav, deg, N);

  embed_kernel<<<ntiles, 32, 0, stream>>>(x, nav, N, emb1.wss, emb1.b, emb1.wsv, dEmb2, sN, vN);

  for (int L = 0; L < 2; ++L) {
    zero_f32<<<grid(64 * N), 256, 0, stream>>>(ags, 64 * N);
    zero_f32<<<grid(96 * N), 256, 0, stream>>>(agv, 96 * N);
    msg_kernel<<<etiles, 32, 0, stream>>>(sN, vN, ei, E, ea, dis, dM1[L], dM2[L], ags, agv);
    upd_kernel<<<ntiles, 32, 0, stream>>>(sN, vN, ags, agv, nav, N, dU1[L], dU2[L]);
  }

  zero_f32<<<grid(16 * 128), 256, 0, stream>>>(gacc, 16 * 128);
  zero_f32<<<1, 256, 0, stream>>>(gcnt, 16);
  head_kernel<<<ntiles, 32, 0, stream>>>(sN, vN, nav, batch, N, dPre1, pre2w16, pre2_b, gacc, gcnt);
  head_final<<<1, 256, 0, stream>>>(gacc, gcnt, post1_w, post1_b, post2_w, post2_b,
                                    (float*)d_out, G);
}
